// LongRangeTemporalSSM_12515534700928
// MI455X (gfx1250) — matrix operand order repaired and checked by reading, not compiled, not run on hardware
//
#include <hip/hip_runtime.h>
#include <hip/hip_bf16.h>
#include <math.h>

#define BB 32
#define TT 4096
#define HH 128
#define NN 128
#define MM (BB*TT)      // 131072 rows

typedef __attribute__((ext_vector_type(16))) __bf16 v16bf;
typedef __attribute__((ext_vector_type(8)))  float  v8f;
typedef __attribute__((ext_vector_type(4)))  unsigned int u32x4;

union Frag16 { v16bf v; u32x4 q[2]; };

__device__ __forceinline__ __bf16 f2bf(float f) {
    union { float f; unsigned u; } a; a.f = f;
    unsigned r = a.u + 0x7FFFu + ((a.u >> 16) & 1u);
    union { unsigned short s; __bf16 h; } b; b.s = (unsigned short)(r >> 16);
    return b.h;
}
__device__ __forceinline__ float bf2f(__bf16 h) {
    union { unsigned short s; __bf16 h; } b; b.h = h;
    union { unsigned u; float f; } a; a.u = ((unsigned)b.s) << 16;
    return a.f;
}

// Load 16 bf16: elements 0..7 from p[0..7], 8..15 from p[16..23]
// (matches A-fragment per-lane chunks; B rows are gathered in the same split k order)
__device__ __forceinline__ v16bf load_frag(const __bf16* p) {
    Frag16 f;
    f.q[0] = *reinterpret_cast<const u32x4*>(p);
    f.q[1] = *reinterpret_cast<const u32x4*>(p + 16);
    return f.v;
}

// ---------------------------------------------------------------- prep kernels

__global__ void __launch_bounds__(256)
k_prep_s5(const float* __restrict__ lam_re, const float* __restrict__ lam_im,
          const float* __restrict__ b_re,   const float* __restrict__ b_im,
          const float* __restrict__ log_step,
          __bf16* __restrict__ Bmat, float2* __restrict__ lamArr)
{
    int i = blockIdx.x * 256 + threadIdx.x;       // d*N*H
    if (i >= 2 * NN * HH) return;
    int h = i & 127, n = (i >> 7) & 127, d = i >> 14;
    int dn = d * NN + n;
    float lr = lam_re[dn], li = lam_im[dn];
    float dt = expf(log_step[dn]);
    float er = expf(lr * dt);
    float cr = er * cosf(li * dt);
    float ci = er * sinf(li * dt);
    float nr = cr - 1.0f, ni = ci;
    float d2 = lr * lr + li * li;
    float qr = (nr * lr + ni * li) / d2;
    float qi = (ni * lr - nr * li) / d2;
    float br = b_re[(size_t)dn * HH + h];
    float bi = b_im[(size_t)dn * HH + h];
    float wr = qr * br - qi * bi;
    float wi = qr * bi + qi * br;
    __bf16* bm = Bmat + (size_t)d * HH * 256;
    bm[h * 256 + n]        = f2bf(wr);
    bm[h * 256 + 128 + n]  = f2bf(wi);
    if (h == 0) lamArr[dn] = make_float2(cr, ci);
}

__global__ void __launch_bounds__(256)
k_prep_C(const float* __restrict__ C_re, const float* __restrict__ C_im,
         __bf16* __restrict__ Cmat)
{
    int i = blockIdx.x * 256 + threadIdx.x;       // [d][256][128]
    if (i >= 2 * 256 * 128) return;
    int h  = i & 127;
    int kk = (i >> 7) & 255;
    int d  = i >> 15;
    float v;
    if (kk < 128) v =  2.0f * C_re[((size_t)(d * HH + h)) * NN + kk];
    else          v = -2.0f * C_im[((size_t)(d * HH + h)) * NN + (kk - 128)];
    Cmat[i] = f2bf(v);
}

__global__ void __launch_bounds__(256)
k_f2bf(const float* __restrict__ s, __bf16* __restrict__ d, int n)
{
    int i = blockIdx.x * 256 + threadIdx.x;
    if (i < n) d[i] = f2bf(s[i]);
}

// Wc[kk=kt*128+h][o] = w[o][h][kt]   (w flat: o*640 + h*5 + kt)
__global__ void __launch_bounds__(256)
k_prep_conv(const float* __restrict__ w, __bf16* __restrict__ dst)
{
    int i = blockIdx.x * 256 + threadIdx.x;       // 640*128
    if (i >= 640 * 128) return;
    int o = i & 127;
    int kk = i >> 7;
    int kt = kk >> 7, h = kk & 127;
    dst[i] = f2bf(w[(size_t)o * 640 + h * 5 + kt]);
}

// ---------------------------------------------------------------- LayerNorm (1 wave / row)
__global__ void __launch_bounds__(256)
k_ln(const float* __restrict__ x, const float* __restrict__ g, const float* __restrict__ be,
     float* __restrict__ xnf, __bf16* __restrict__ xnb)
{
    int lane = threadIdx.x & 31;
    int row  = blockIdx.x * 8 + (threadIdx.x >> 5);
    const float4 v = *reinterpret_cast<const float4*>(x + (size_t)row * HH + lane * 4);
    float s = v.x + v.y + v.z + v.w;
    #pragma unroll
    for (int m = 16; m >= 1; m >>= 1) s += __shfl_xor(s, m, 32);
    float mu = s * (1.0f / 128.0f);
    float d0 = v.x - mu, d1 = v.y - mu, d2 = v.z - mu, d3 = v.w - mu;
    float q = d0 * d0 + d1 * d1 + d2 * d2 + d3 * d3;
    #pragma unroll
    for (int m = 16; m >= 1; m >>= 1) q += __shfl_xor(q, m, 32);
    float rs = rsqrtf(q * (1.0f / 128.0f) + 1e-5f);
    int c = lane * 4;
    float o0 = d0 * rs * g[c + 0] + be[c + 0];
    float o1 = d1 * rs * g[c + 1] + be[c + 1];
    float o2 = d2 * rs * g[c + 2] + be[c + 2];
    float o3 = d3 * rs * g[c + 3] + be[c + 3];
    *reinterpret_cast<float4*>(xnf + (size_t)row * HH + c) = make_float4(o0, o1, o2, o3);
    union { __bf16 h[4]; unsigned long long u; } pk;
    pk.h[0] = f2bf(o0); pk.h[1] = f2bf(o1); pk.h[2] = f2bf(o2); pk.h[3] = f2bf(o3);
    *reinterpret_cast<unsigned long long*>(xnb + (size_t)row * HH + c) = pk.u;
}

// ---------------------------------------------------------------- WMMA GEMM
// A [M][K] bf16 row-major, Bw [K][LDN] bf16 row-major; block stages its 128-col
// B strip in LDS once, then 8 waves each compute a 16x128 strip of D.
// MODE 0: store bf16 (Bu)          MODE 1: +resid +xn*dskip -> f32 (S5 out)
// MODE 2: +bias, GELU -> bf16      MODE 3: +resid +bias -> f32 and bf16 (MLP out)
template<int K, int LDN, int LDO, int MODE>
__global__ void __launch_bounds__(256)
k_gemm(const __bf16* __restrict__ A, const __bf16* __restrict__ Bw,
       const float* __restrict__ bias, const float* __restrict__ resid,
       const float* __restrict__ xnf,  const float* __restrict__ dskip,
       float* __restrict__ outf, __bf16* __restrict__ outb)
{
    __shared__ __bf16 Blds[K * 128];
    const int lane = threadIdx.x & 31;
    const int wave = threadIdx.x >> 5;
    const int m0 = (blockIdx.x * 8 + wave) * 16;
    const int n0 = blockIdx.y * 128;

    // cooperative stage of the B strip: rows 0..K-1, cols n0..n0+127
    {
        constexpr int CH = (K * 128) / (256 * 8);    // 16B chunks per thread
        #pragma unroll
        for (int c = 0; c < CH; ++c) {
            int idx = (c * 256 + (int)threadIdx.x) * 8;
            int row = idx >> 7;
            int col = idx & 127;
            *reinterpret_cast<u32x4*>(&Blds[idx]) =
                *reinterpret_cast<const u32x4*>(Bw + (size_t)row * LDN + n0 + col);
        }
    }
    __syncthreads();

    const int kb = (lane & 16) ? 8 : 0;
    const __bf16* arow = A + (size_t)(m0 + (lane & 15)) * K;
    v8f zero = {0.f, 0.f, 0.f, 0.f, 0.f, 0.f, 0.f, 0.f};
    v8f acc[8];
    #pragma unroll
    for (int j = 0; j < 8; ++j) acc[j] = zero;
    #pragma unroll
    for (int s = 0; s < (K >> 5); ++s) {
        v16bf af = load_frag(arow + s * 32 + kb);
        const __bf16* bbase = &Blds[(s * 32 + kb) * 128 + (lane & 15)];
        #pragma unroll
        for (int j = 0; j < 8; ++j) {
            v16bf bfv;
            #pragma unroll
            for (int e = 0; e < 16; ++e) bfv[e] = bbase[((e & 7) + ((e >> 3) << 4)) * 128 + j * 16];
            acc[j] = __builtin_amdgcn_wmma_f32_16x16x32_bf16(
                false, af, false, bfv, (short)0, acc[j], false, false);
        }
    }
    const int cb = n0 + (lane & 15);
    const int rb = m0 + ((lane & 16) ? 8 : 0);
    #pragma unroll
    for (int j = 0; j < 8; ++j) {
        int col = cb + j * 16;
        #pragma unroll
        for (int r = 0; r < 8; ++r) {
            int row = rb + r;
            float v = acc[j][r];
            if (MODE == 0) {
                outb[(size_t)row * LDO + col] = f2bf(v);
            } else if (MODE == 1) {
                v += resid[(size_t)row * HH + col] + xnf[(size_t)row * HH + col] * dskip[col];
                outf[(size_t)row * LDO + col] = v;
            } else if (MODE == 2) {
                v += bias[col];
                v = 0.5f * v * (1.0f + erff(v * 0.70710678118654752f));
                outb[(size_t)row * LDO + col] = f2bf(v);
            } else {
                v += resid[(size_t)row * HH + col] + bias[col];
                outf[(size_t)row * LDO + col] = v;
                outb[(size_t)row * HH + col]  = f2bf(v);
            }
        }
    }
}

// ---------------------------------------------------------------- S5 scan (1 lane / (b,n) chain)
__global__ void __launch_bounds__(256)
k_scan(const __bf16* __restrict__ Bu, const float2* __restrict__ lamArr,
       __bf16* __restrict__ xs)
{
    int c = blockIdx.x * 256 + threadIdx.x;       // 0..4095
    int b = c >> 7, n = c & 127;
    float2 a = lamArr[n];
    float xr = 0.0f, xi = 0.0f;
    const __bf16* pi = Bu + ((size_t)b * TT) * 256 + n;
    __bf16* po = xs + ((size_t)b * TT) * 256 + n;
    for (int t = 0; t < TT; ++t) {
        float br = bf2f(pi[0]);
        float bi = bf2f(pi[128]);
        float nr = a.x * xr - a.y * xi + br;
        float ni = a.x * xi + a.y * xr + bi;
        xr = nr; xi = ni;
        po[0]   = f2bf(xr);
        po[128] = f2bf(xi);
        pi += 256; po += 256;
    }
}

// ---------------------------------------------------------------- conv as implicit-im2col WMMA GEMM
// xin [B][TIN][128] bf16, Wc [640][128] bf16, yout [B][TOUT][128] bf16, ReLU fused.
template<int TIN, int TOUT>
__global__ void __launch_bounds__(256)
k_conv(const __bf16* __restrict__ xin, const __bf16* __restrict__ Wc,
       const float* __restrict__ bias, __bf16* __restrict__ yout)
{
    const int lane = threadIdx.x & 31;
    const int m0 = (blockIdx.x * 8 + (threadIdx.x >> 5)) * 16;
    const int b = m0 / TOUT;
    const int o = (m0 % TOUT) + (lane & 15);
    const int kb = (lane & 16) ? 8 : 0;
    v8f zero = {0.f, 0.f, 0.f, 0.f, 0.f, 0.f, 0.f, 0.f};
    v8f acc[8];
    #pragma unroll
    for (int j = 0; j < 8; ++j) acc[j] = zero;
    #pragma unroll 4
    for (int s = 0; s < 20; ++s) {                // K = 640 = 20*32
        Frag16 fa;
        int kkA = s * 32 + kb;
        #pragma unroll
        for (int cchunk = 0; cchunk < 2; ++cchunk) {
            int kk = kkA + cchunk * 16;
            int kt = kk >> 7;
            int h  = kk & 127;
            int t  = 2 * o - 2 + kt;
            u32x4 z = {0u, 0u, 0u, 0u};
            if ((unsigned)t < (unsigned)TIN)
                z = *reinterpret_cast<const u32x4*>(xin + (((size_t)(b * TIN + t)) << 7) + h);
            fa.q[cchunk] = z;
        }
        const __bf16* bbase = Wc + (size_t)(s * 32 + kb) * 128 + (lane & 15);
        #pragma unroll
        for (int j = 0; j < 8; ++j) {
            v16bf bfv;
            #pragma unroll
            for (int e = 0; e < 16; ++e) bfv[e] = bbase[((e & 7) + ((e >> 3) << 4)) * 128 + j * 16];
            acc[j] = __builtin_amdgcn_wmma_f32_16x16x32_bf16(
                false, fa.v, false, bfv, (short)0, acc[j], false, false);
        }
    }
    const int cb = lane & 15;
    const int rb = m0 + ((lane & 16) ? 8 : 0);
    #pragma unroll
    for (int j = 0; j < 8; ++j) {
        int col = cb + j * 16;
        #pragma unroll
        for (int r = 0; r < 8; ++r) {
            int row = rb + r;
            float v = acc[j][r] + bias[col];
            v = fmaxf(v, 0.0f);
            yout[(size_t)row * 128 + col] = f2bf(v);
        }
    }
}

// ---------------------------------------------------------------- adaptive avg pool
__global__ void __launch_bounds__(256)
k_pool(const __bf16* __restrict__ c2, float* __restrict__ out)
{
    int i = blockIdx.x * 256 + threadIdx.x;       // B*128*64
    if (i >= BB * HH * 64) return;
    int d = i & 63;
    int h = (i >> 6) & 127;
    int b = i >> 13;
    const __bf16* p = c2 + ((size_t)(b * 1024 + d * 16)) * 128 + h;
    float s = 0.0f;
    #pragma unroll
    for (int j = 0; j < 16; ++j) { s += bf2f(*p); p += 128; }
    out[i] = s * (1.0f / 16.0f);
}

// ---------------------------------------------------------------- launch
extern "C" void kernel_launch(void* const* d_in, const int* in_sizes, int n_in,
                              void* d_out, int out_size, void* d_ws, size_t ws_size,
                              hipStream_t stream) {
    (void)in_sizes; (void)n_in; (void)out_size; (void)ws_size;
    const float* x_in    = (const float*)d_in[0];
    const float* ln1_g   = (const float*)d_in[1];
    const float* ln1_b   = (const float*)d_in[2];
    const float* ln2_g   = (const float*)d_in[3];
    const float* ln2_b   = (const float*)d_in[4];
    const float* lam_re  = (const float*)d_in[5];
    const float* lam_im  = (const float*)d_in[6];
    const float* B_re    = (const float*)d_in[7];
    const float* B_im    = (const float*)d_in[8];
    const float* C_re    = (const float*)d_in[9];
    const float* C_im    = (const float*)d_in[10];
    const float* D_skip  = (const float*)d_in[11];
    const float* log_st  = (const float*)d_in[12];
    const float* W1      = (const float*)d_in[13];
    const float* b1      = (const float*)d_in[14];
    const float* W2      = (const float*)d_in[15];
    const float* b2      = (const float*)d_in[16];
    const float* c1w     = (const float*)d_in[17];
    const float* c1b     = (const float*)d_in[18];
    const float* c2w     = (const float*)d_in[19];
    const float* c2b     = (const float*)d_in[20];
    float* out = (float*)d_out;

    char* ws = (char*)d_ws;
    size_t off = 0;
    auto alloc = [&](size_t bytes) -> char* {
        char* p = ws + off;
        off += (bytes + 255) & ~(size_t)255;
        return p;
    };
    float*  xbuf = (float*)alloc((size_t)MM * HH * 4);     // running fp32 state
    float*  xnf  = (float*)alloc((size_t)MM * HH * 4);     // LN out fp32
    __bf16* xnb  = (__bf16*)alloc((size_t)MM * HH * 2);    // LN out bf16
    __bf16* bu   = (__bf16*)alloc((size_t)MM * 256 * 2);   // Bu (re|im), reused as MLP hidden
    __bf16* xsb  = (__bf16*)alloc((size_t)MM * 256 * 2);   // scan states (re|im) bf16
    __bf16* xb   = (__bf16*)alloc((size_t)MM * HH * 2);    // bf16 copy of state for conv
    __bf16* c1   = (__bf16*)alloc((size_t)BB * 2048 * 128 * 2);
    __bf16* c2   = (__bf16*)alloc((size_t)BB * 1024 * 128 * 2);
    __bf16* Bmat = (__bf16*)alloc((size_t)2 * 128 * 256 * 2);
    float2* lamA = (float2*)alloc((size_t)2 * 128 * sizeof(float2));
    __bf16* Cmat = (__bf16*)alloc((size_t)2 * 256 * 128 * 2);
    __bf16* W1t  = (__bf16*)alloc((size_t)2 * 128 * 256 * 2);
    __bf16* W2t  = (__bf16*)alloc((size_t)2 * 256 * 128 * 2);
    __bf16* Wc1  = (__bf16*)alloc((size_t)640 * 128 * 2);
    __bf16* Wc2  = (__bf16*)alloc((size_t)640 * 128 * 2);

    dim3 blk(256);
    k_prep_s5<<<128, blk, 0, stream>>>(lam_re, lam_im, B_re, B_im, log_st, Bmat, lamA);
    k_prep_C<<<256, blk, 0, stream>>>(C_re, C_im, Cmat);
    k_f2bf<<<256, blk, 0, stream>>>(W1, W1t, 2 * 128 * 256);
    k_f2bf<<<256, blk, 0, stream>>>(W2, W2t, 2 * 256 * 128);
    k_prep_conv<<<320, blk, 0, stream>>>(c1w, Wc1);
    k_prep_conv<<<320, blk, 0, stream>>>(c2w, Wc2);

    for (int d = 0; d < 2; ++d) {
        const float* xstate = (d == 0) ? x_in : xbuf;
        // S5 sub-block
        k_ln<<<MM / 8, blk, 0, stream>>>(xstate, ln1_g + d * 128, ln1_b + d * 128, xnf, xnb);
        k_gemm<128, 256, 256, 0><<<dim3(MM / 128, 2), blk, 0, stream>>>(
            xnb, Bmat + (size_t)d * 128 * 256, nullptr, nullptr, nullptr, nullptr, nullptr, bu);
        k_scan<<<16, blk, 0, stream>>>(bu, lamA + d * 128, xsb);
        k_gemm<256, 128, 128, 1><<<dim3(MM / 128, 1), blk, 0, stream>>>(
            xsb, Cmat + (size_t)d * 256 * 128, nullptr, xstate, xnf, D_skip + d * 128, xbuf, nullptr);
        // MLP sub-block
        k_ln<<<MM / 8, blk, 0, stream>>>(xbuf, ln2_g + d * 128, ln2_b + d * 128, xnf, xnb);
        k_gemm<128, 256, 256, 2><<<dim3(MM / 128, 2), blk, 0, stream>>>(
            xnb, W1t + (size_t)d * 128 * 256, b1 + d * 256, nullptr, nullptr, nullptr, nullptr, bu);
        k_gemm<256, 128, 128, 3><<<dim3(MM / 128, 1), blk, 0, stream>>>(
            bu, W2t + (size_t)d * 256 * 128, b2 + d * 128, xbuf, nullptr, nullptr, xbuf, xb);
    }

    k_conv<4096, 2048><<<dim3((BB * 2048) / 128, 1), blk, 0, stream>>>(xb, Wc1, c1b, c1);
    k_conv<2048, 1024><<<dim3((BB * 1024) / 128, 1), blk, 0, stream>>>(c1, Wc2, c2b, c2);
    k_pool<<<1024, blk, 0, stream>>>(c2, out);
}
